// MultiHeadAttention_996432413290
// MI455X (gfx1250) — matrix operand order repaired and checked by reading, not compiled, not run on hardware
//
#include <hip/hip_runtime.h>
#include <hip/hip_bf16.h>

typedef __attribute__((ext_vector_type(16))) _Float16 v16h;
typedef __attribute__((ext_vector_type(8)))  _Float16 v8h;
typedef __attribute__((ext_vector_type(8)))  float    v8f;

#define SEQ   2048
#define HDIM  512
#define DKD   64
#define NHEAD 8
#define NB    2
#define MROWS 4096          // B*S
#define QKSCALE 0.125f      // 1/sqrt(64)

union frag_u { v16h v; v8h h[2]; };

__device__ __forceinline__ v8f wmma16(v16h a, v16h b, v8f c) {
  // emits v_wmma_f32_16x16x32_f16
  return __builtin_amdgcn_wmma_f32_16x16x32_f16(false, a, false, b, (short)0, c, false, false);
}

// A-fragment 16x32 f16 from row-major [row][k] storage.
// Per ISA: lanes 0-15 -> K 0..7 & 16..23, lanes 16-31 -> K 8..15 & 24..31.
__device__ __forceinline__ v16h frag_a(const _Float16* base, int stride, int lane) {
  const int row = lane & 15;
  const int kb  = (lane >> 4) * 8;
  const _Float16* p = base + row * stride + kb;
  frag_u u;
  u.h[0] = *(const v8h*)(p);
  u.h[1] = *(const v8h*)(p + 16);
  return u.v;
}

// B-fragment 32x16 f16 from "transposed" storage T[n][k] row-major.
// Per ISA: lanes 0-15 -> K 0..7 & 16..23, lanes 16-31 -> K 8..15 & 24..31, column n = lane&15.
__device__ __forceinline__ v16h frag_b_t(const _Float16* baseT, int stride, int lane) {
  const int n  = lane & 15;
  const int kb = (lane >> 4) * 8;
  const _Float16* p = baseT + n * stride + kb;
  frag_u u;
  u.h[0] = *(const v8h*)(p);
  u.h[1] = *(const v8h*)(p + 16);
  return u.v;
}

__device__ __forceinline__ float rowmax16(float v) {
#pragma unroll
  for (int off = 8; off > 0; off >>= 1) v = fmaxf(v, __shfl_xor(v, off, 16));
  return v;
}
__device__ __forceinline__ float rowsum16(float v) {
#pragma unroll
  for (int off = 8; off > 0; off >>= 1) v += __shfl_xor(v, off, 16);
  return v;
}

// ---------------- cast f32 -> f16 ----------------
__global__ __launch_bounds__(256) void cast_f16_kernel(const float* __restrict__ in,
                                                       _Float16* __restrict__ out, int n) {
  int i = blockIdx.x * 256 + threadIdx.x;
  if (i < n) out[i] = (_Float16)in[i];
}

// cast + transpose: in [rows][cols] f32 -> out [cols][rows] f16
__global__ __launch_bounds__(256) void cast_t_f16_kernel(const float* __restrict__ in,
                                                         _Float16* __restrict__ outT,
                                                         int rows, int cols) {
  int i = blockIdx.x * 256 + threadIdx.x;
  if (i < rows * cols) {
    int r = i / cols, c = i - r * cols;
    outT[(size_t)c * rows + r] = (_Float16)in[i];
  }
}

// ---------------- GEMM: C[M,512] = A[M,512] * W + bias  (W given TRANSPOSED [n][k]) ----
// mode 0: f16 head-split [(b*8+h)*2048 + s]*64 + dk              (Q, K)
// mode 2: f16 head-split TRANSPOSED [((b*8+h)*64 + dk)*2048 + s] (V)
// mode 1: f32 plain [m*512 + n]                                   (final output)
__global__ __launch_bounds__(256) void gemm_kernel(const _Float16* __restrict__ A,
                                                   const _Float16* __restrict__ Wt,
                                                   const float* __restrict__ bias,
                                                   _Float16* __restrict__ out16,
                                                   float* __restrict__ out32,
                                                   int mode) {
  const int lane = threadIdx.x & 31;
  const int w    = threadIdx.x >> 5;                  // 8 waves: 2(M) x 4(N)
  const int m0   = blockIdx.x * 32 + (w >> 2) * 16;
  const int n0   = blockIdx.y * 64 + (w & 3) * 16;

  v8f acc = {};
#pragma unroll
  for (int kk = 0; kk < HDIM; kk += 32) {
    v16h a = frag_a(A + (size_t)m0 * HDIM + kk, HDIM, lane);
    v16h b = frag_b_t(Wt + (size_t)n0 * HDIM + kk, HDIM, lane);
    acc = wmma16(a, b, acc);
  }

  const int nIdx = lane & 15;
  const int half = lane >> 4;
  const float bn = bias[n0 + nIdx];
#pragma unroll
  for (int r = 0; r < 8; ++r) {
    const float v = acc[r] + bn;
    const int m = m0 + r + 8 * half;
    const int n = n0 + nIdx;
    const int b = m >> 11, s = m & (SEQ - 1);
    const int h = n >> 6, dk = n & 63;
    if (mode == 0) {
      out16[(((size_t)(b * NHEAD + h) * SEQ) + s) * DKD + dk] = (_Float16)v;
    } else if (mode == 2) {
      out16[(((size_t)(b * NHEAD + h) * DKD) + dk) * SEQ + s] = (_Float16)v;
    } else {
      out32[(size_t)m * HDIM + n] = v;
    }
  }
}

// ---------------- Flash attention (barrier-free) ----------------
// Qh/Kh: [BH][S][64] f16 row-major. Vt: [BH][64][S] f16 (transposed).
// bias: [BH][S][S] f32. Ctx: [B*S][512] f16 (heads concat).
__global__ __launch_bounds__(256) void attn_kernel(const _Float16* __restrict__ Qh,
                                                   const _Float16* __restrict__ Kh,
                                                   const _Float16* __restrict__ Vt,
                                                   const float* __restrict__ bias,
                                                   _Float16* __restrict__ Ctx) {
  __shared__ _Float16 Pt[8][16 * 32];     // per-wave exp'd probs [row][k]

  const int tid  = threadIdx.x;
  const int lane = tid & 31;
  const int w    = tid >> 5;
  const int bh   = blockIdx.y;                      // 0..15
  const int q0   = blockIdx.x * 128 + w * 16;       // q-row tile per wave
  const int nIdx = lane & 15;
  const int half = lane >> 4;

  const _Float16* qb  = Qh + (size_t)bh * SEQ * DKD;
  const _Float16* kb  = Kh + (size_t)bh * SEQ * DKD;   // [s][dk]
  const _Float16* vtb = Vt + (size_t)bh * DKD * SEQ;   // [dk][s]
  const float*    bb  = bias + (size_t)bh * SEQ * SEQ;

  // Q fragments over the two 32-wide dk chunks
  const v16h aq0 = frag_a(qb + (size_t)q0 * DKD + 0,  DKD, lane);
  const v16h aq1 = frag_a(qb + (size_t)q0 * DKD + 32, DKD, lane);

  v8f o0 = {}, o1 = {}, o2 = {}, o3 = {};
  float mrow[8], lrow[8];
#pragma unroll
  for (int r = 0; r < 8; ++r) { mrow[r] = -3.0e38f; lrow[r] = 0.0f; }

  _Float16* pt = Pt[w];

  for (int j = 0; j < SEQ; j += 32) {
    if (j + 32 < SEQ) { // prefetch next K rows / V columns
      __builtin_prefetch(kb + (size_t)(j + 32 + lane) * DKD, 0, 1);
      __builtin_prefetch(vtb + (size_t)lane * SEQ + j + 32, 0, 1);
      __builtin_prefetch(vtb + (size_t)(lane + 32) * SEQ + j + 32, 0, 1);
    }

    // scores: two 16-key sub-tiles, each accumulated over dk=64 (2 WMMAs),
    // B-fragments straight from global (dk contiguous per key row)
    v8f s0 = {}, s1 = {};
    s0 = wmma16(aq0, frag_b_t(kb + (size_t)(j +  0) * DKD + 0,  DKD, lane), s0);
    s0 = wmma16(aq1, frag_b_t(kb + (size_t)(j +  0) * DKD + 32, DKD, lane), s0);
    s1 = wmma16(aq0, frag_b_t(kb + (size_t)(j + 16) * DKD + 0,  DKD, lane), s1);
    s1 = wmma16(aq1, frag_b_t(kb + (size_t)(j + 16) * DKD + 32, DKD, lane), s1);

#pragma unroll
    for (int r = 0; r < 8; ++r) {
      const int m = r + 8 * half;                     // row within q-tile
      const size_t brow = (size_t)(q0 + m) * SEQ + j;
      // bias is streamed once (268MB > L2): non-temporal loads
      float a0 = s0[r] * QKSCALE + __builtin_nontemporal_load(&bb[brow + nIdx]);
      float a1 = s1[r] * QKSCALE + __builtin_nontemporal_load(&bb[brow + 16 + nIdx]);
      const float mcur = rowmax16(fmaxf(a0, a1));
      const float mnew = fmaxf(mrow[r], mcur);
      const float corr = __expf(mrow[r] - mnew);
      mrow[r] = mnew;
      const float p0 = __expf(a0 - mnew);
      const float p1 = __expf(a1 - mnew);
      lrow[r] = lrow[r] * corr + rowsum16(p0 + p1);
      o0[r] *= corr; o1[r] *= corr; o2[r] *= corr; o3[r] *= corr;
      pt[m * 32 + nIdx]      = (_Float16)p0;
      pt[m * 32 + 16 + nIdx] = (_Float16)p1;
    }

    // P (16x32) as A-fragment via per-wave LDS reshape; V B-frags direct from
    // transposed global layout (kv contiguous per dk row)
    const v16h ap = frag_a(pt, 32, lane);
    o0 = wmma16(ap, frag_b_t(vtb + (size_t) 0 * SEQ + j, SEQ, lane), o0);
    o1 = wmma16(ap, frag_b_t(vtb + (size_t)16 * SEQ + j, SEQ, lane), o1);
    o2 = wmma16(ap, frag_b_t(vtb + (size_t)32 * SEQ + j, SEQ, lane), o2);
    o3 = wmma16(ap, frag_b_t(vtb + (size_t)48 * SEQ + j, SEQ, lane), o3);
  }

  // normalize and store ctx in [B*S][512] heads-concat f16 layout
  const int b = bh >> 3, h = bh & 7;
#pragma unroll
  for (int r = 0; r < 8; ++r) {
    const float inv = 1.0f / lrow[r];
    const int s = q0 + r + 8 * half;
    _Float16* dst = Ctx + ((size_t)(b * SEQ + s)) * HDIM + h * DKD + nIdx;
    dst[0]  = (_Float16)(o0[r] * inv);
    dst[16] = (_Float16)(o1[r] * inv);
    dst[32] = (_Float16)(o2[r] * inv);
    dst[48] = (_Float16)(o3[r] * inv);
  }
}

extern "C" void kernel_launch(void* const* d_in, const int* in_sizes, int n_in,
                              void* d_out, int out_size, void* d_ws, size_t ws_size,
                              hipStream_t stream) {
  const float* q  = (const float*)d_in[0];
  const float* k  = (const float*)d_in[1];
  const float* v  = (const float*)d_in[2];
  const float* ab = (const float*)d_in[3];
  const float* Wq = (const float*)d_in[4];
  const float* bq = (const float*)d_in[5];
  const float* Wk = (const float*)d_in[6];
  const float* bk = (const float*)d_in[7];
  const float* Wv = (const float*)d_in[8];
  const float* bv = (const float*)d_in[9];
  const float* Wo = (const float*)d_in[10];
  const float* bo = (const float*)d_in[11];
  float* out = (float*)d_out;

  const size_t NE = (size_t)MROWS * HDIM;   // 2M elems
  const size_t NW = (size_t)HDIM * HDIM;    // 256K elems
  _Float16* p = (_Float16*)d_ws;
  _Float16 *Q16 = p;            p += NE;
  _Float16 *K16 = p;            p += NE;
  _Float16 *V16 = p;            p += NE;
  _Float16 *WqT = p;            p += NW;
  _Float16 *WkT = p;            p += NW;
  _Float16 *WvT = p;            p += NW;
  _Float16 *WoT = p;            p += NW;
  _Float16 *Qhd = p;            p += NE;
  _Float16 *Khd = p;            p += NE;
  _Float16 *Vtr = p;            p += NE;
  _Float16 *Ctx16 = p;          p += NE;

  // stage 1: f32 -> f16 casts (weights also transposed to [n][k])
  cast_f16_kernel<<<dim3((int)(NE / 256)), 256, 0, stream>>>(q, Q16, (int)NE);
  cast_f16_kernel<<<dim3((int)(NE / 256)), 256, 0, stream>>>(k, K16, (int)NE);
  cast_f16_kernel<<<dim3((int)(NE / 256)), 256, 0, stream>>>(v, V16, (int)NE);
  cast_t_f16_kernel<<<dim3((int)(NW / 256)), 256, 0, stream>>>(Wq, WqT, HDIM, HDIM);
  cast_t_f16_kernel<<<dim3((int)(NW / 256)), 256, 0, stream>>>(Wk, WkT, HDIM, HDIM);
  cast_t_f16_kernel<<<dim3((int)(NW / 256)), 256, 0, stream>>>(Wv, WvT, HDIM, HDIM);
  cast_t_f16_kernel<<<dim3((int)(NW / 256)), 256, 0, stream>>>(Wo, WoT, HDIM, HDIM);

  // stage 2: QKV projections (Q,K row-major heads; V transposed heads)
  dim3 ggrid(MROWS / 32, HDIM / 64);
  gemm_kernel<<<ggrid, 256, 0, stream>>>(Q16, WqT, bq, Qhd, nullptr, 0);
  gemm_kernel<<<ggrid, 256, 0, stream>>>(K16, WkT, bk, Khd, nullptr, 0);
  gemm_kernel<<<ggrid, 256, 0, stream>>>(V16, WvT, bv, Vtr, nullptr, 2);

  // stage 3: flash attention with additive bias (no block barriers)
  attn_kernel<<<dim3(SEQ / 128, NB * NHEAD), 256, 0, stream>>>(Qhd, Khd, Vtr, ab, Ctx16);

  // stage 4: output projection -> f32 d_out
  gemm_kernel<<<ggrid, 256, 0, stream>>>(Ctx16, WoT, bo, nullptr, out, 1);
}
